// SpectralTransform_38457137168994
// MI455X (gfx1250) — matrix operand order repaired and checked by reading, not compiled, not run on hardware
//
#include <hip/hip_runtime.h>
#include <math.h>

// CDNA5 (gfx1250) wave32 WMMA fragments
typedef __attribute__((ext_vector_type(16))) _Float16 v16h;
typedef __attribute__((ext_vector_type(8)))  _Float16 v8h;
typedef __attribute__((ext_vector_type(8)))  float    v8f;
typedef __attribute__((ext_vector_type(2)))  float    v2f;

#define B_      2
#define NVOX    110592              // 48^3
#define CIN     16
#define COUT    16
#define FDIM    256
#define KCHUNK  32                  // K of v_wmma_f32_16x16x32_f16
#define NCHUNKS (NVOX / KCHUNK)     // 3456 per batch
#define NTILES  (B_ * (FDIM / 16))  // 32 output tiles (b, f-tile)
#define NCH2    (B_ * NCHUNKS)      // 6912 chunks over both batches
#define A_SW_ELEMS ((size_t)NCH2 * 512)          // f16 elements (3,538,944)
#define A_SW_BYTES (A_SW_ELEMS * 2)              // 6.75 MB, 256-B aligned

// ---------------------------------------------------------------------------
// Kernel 0 (pre-pass): convert a to f16, swizzled into B-fragment register
// order: a_sw[c*512 + lane*16 + j] = (f16) a[c*32 + 8*(lane>>4) + j + (j&8), lane&15]
// so the main loop fetches a whole 32x16 B-fragment as one v16h per lane.
// ---------------------------------------------------------------------------
__global__ __launch_bounds__(256)
void swizzle_a_f16(const float* __restrict__ a, _Float16* __restrict__ a_sw)
{
    const size_t t = (size_t)blockIdx.x * blockDim.x + threadIdx.x; // octet id
    if (t >= A_SW_ELEMS / 8) return;
    const size_t o  = t * 8;                 // first f16 element of this octet
    const int    c  = (int)(o >> 9);         // chunk (512 f16 per chunk)
    const int    ln = (int)((o >> 4) & 31);  // lane slot
    const int    j0 = (int)(o & 15);         // 0 or 8
    const int    i  = ln & 15;
    const int    lh = ln >> 4;

    v8h v;
    #pragma unroll
    for (int jj = 0; jj < 8; ++jj) {
        const size_t n = (size_t)c * 32 + lh * 8 + 2 * j0 + jj; // flat voxel id
        v[jj] = (_Float16)a[n * CIN + i];
    }
    *(v8h*)(a_sw + o) = v;                   // 16-B coalesced store
}

// ---------------------------------------------------------------------------
// Kernel 1: per (tile, k-slice) wave computes partial
//   G_re[f,i] = sum_n cos(2pi x.w_f) * a[n,i]
//   G_im[f,i] = sum_n sin(2pi x.w_f) * a[n,i]
//
// Phase GEMM on the matrix pipe (V_WMMA_F32_16X16X4_F32): D hands lane
// (fl,lh) exactly the K-set of the identity-mapped f16 A-fragment, so
// cos/sin (v_cos/v_sin compute cos/sin(2pi*src) natively) apply lane-locally.
// The A-fragment's K=3 element is a don't-care because the B-fragment's K=3
// row is zeroed, so all x loads are unconditional (no exec-mask branches).
// Wave-uniform state is scalarized via readfirstlane.
// ---------------------------------------------------------------------------
template <int USE_SWZ>
__global__ __launch_bounds__(256)
void spectral_wmma_partial(const float* __restrict__ a,
                           const _Float16* __restrict__ a_sw,
                           const float* __restrict__ x,
                           const float* __restrict__ xw,   // x_weights (3x256)
                           float* __restrict__ ws,
                           int S, int cps)
{
    const int lane = threadIdx.x & 31;
    // wave-uniform by construction; make it explicit for SALU codegen
    const int wid = __builtin_amdgcn_readfirstlane(
        (int)((blockIdx.x * blockDim.x + threadIdx.x) >> 5));
    if (wid >= NTILES * S) return;          // scalar test: EXEC stays all-ones

    const int slice = wid % S;
    const int tile  = wid / S;
    const int b     = tile >> 4;            // batch
    const int ft    = tile & 15;            // f-tile (16 freqs)

    const int fl = lane & 15;               // f col / i col / x row selector
    const int lh = lane >> 4;               // lane half

    // B-fragment (4x16 f32) of the phase WMMA = x_weights columns, K=3 row
    // zero-padded. 32-bit B 4x16 layout (mirror of A 16x4): VGPR0 holds K=0
    // (lanes 0-15) / K=2 (lanes 16-31); VGPR1 holds K=1 / K=3(=0).
    const float xw1 = xw[1 * FDIM + ft * 16 + fl];   // unconditional load
    v2f xwb;
    xwb[0] = xw[(lh ? 2 : 0) * FDIM + ft * 16 + fl];
    xwb[1] = lh ? 0.0f : xw1;                        // v_cndmask, one-time

    const v8f zero = {};
    v8f acc_re = {};
    v8f acc_im = {};

    const int c0 = slice * cps;
    int c1 = c0 + cps;
    if (c1 > NCHUNKS) c1 = NCHUNKS;

    // scalar bases; per-lane offsets are loop-invariant
    const float*    x_b  = x + (size_t)b * NVOX * 3;
    const _Float16* sw_b = a_sw + (size_t)b * NCHUNKS * 512 + lane * 16;
    const float*    a_b  = a + ((size_t)b * NVOX + 8 * lh) * CIN + fl;

    for (int ch = c0; ch < c1; ++ch) {
        const int nbase = ch * KCHUNK;

        // x A-fragments (16x4 f32): VGPR0 = K=0 (lanes 0-15) / K=2 (16-31);
        // VGPR1 = K=1 (lanes 0-15) / K=3 don't-care (16-31). All loads
        // unconditional: lane's voxel row = fl.
        const float* xp0 = x_b + (size_t)nbase * 3;
        const float* xp1 = xp0 + 48;        // second 16-voxel block
        v2f xa0, xa1;
        xa0[0] = xp0[fl * 3 + 2 * lh];
        xa0[1] = xp0[fl * 3 + 1];
        xa1[0] = xp1[fl * 3 + 2 * lh];
        xa1[1] = xp1[fl * 3 + 1];

        // phases (divided by 2*pi): D[n = r+8lh, f = fl]
        v8f ph0 = __builtin_amdgcn_wmma_f32_16x16x4_f32(
            false, xa0, false, xwb, (short)0, zero, false, false);
        v8f ph1 = __builtin_amdgcn_wmma_f32_16x16x4_f32(
            false, xa1, false, xwb, (short)0, zero, false, false);

        // B-fragment (32x16 f16), identity K map.
        v16h bf;
        if (USE_SWZ) {
            // pre-swizzled: entire fragment = 32 contiguous bytes per lane
            bf = *(const v16h*)(sw_b + (size_t)ch * 512);
        } else {
            // fallback: element j -> voxel nbase + j + (j&8) + 8*lh, channel fl
            const float* ap = a_b + (size_t)nbase * CIN;
            #pragma unroll
            for (int j = 0; j < 16; ++j)
                bf[j] = (_Float16)ap[(size_t)(j + (j & 8)) * CIN];
        }

        // cos/sin lane-locally into the f16 A-fragment (identity K map:
        // element j < 8 <- block0[j], element j >= 8 <- block1[j-8]).
        v16h cf, sf;
        #pragma unroll
        for (int j = 0; j < 8; ++j) {
#if __has_builtin(__builtin_amdgcn_cosf) && __has_builtin(__builtin_amdgcn_sinf)
            cf[j]     = (_Float16)__builtin_amdgcn_cosf(ph0[j]);
            sf[j]     = (_Float16)__builtin_amdgcn_sinf(ph0[j]);
            cf[j + 8] = (_Float16)__builtin_amdgcn_cosf(ph1[j]);
            sf[j + 8] = (_Float16)__builtin_amdgcn_sinf(ph1[j]);
#else
            cf[j]     = (_Float16)__cosf(6.2831853071795864769f * ph0[j]);
            sf[j]     = (_Float16)__sinf(6.2831853071795864769f * ph0[j]);
            cf[j + 8] = (_Float16)__cosf(6.2831853071795864769f * ph1[j]);
            sf[j + 8] = (_Float16)__sinf(6.2831853071795864769f * ph1[j]);
#endif
        }

        acc_re = __builtin_amdgcn_wmma_f32_16x16x32_f16(
            false, cf, false, bf, (short)0, acc_re, false, false);
        acc_im = __builtin_amdgcn_wmma_f32_16x16x32_f16(
            false, sf, false, bf, (short)0, acc_im, false, false);
    }

    // D layout: VGPR r, lane -> G[f = r + 8*lh, i = fl]; coalesced stores
    float* out = ws + (size_t)(tile * S + slice) * 2 * 256;
    #pragma unroll
    for (int r = 0; r < 8; ++r) {
        out[(r + 8 * lh) * 16 + fl]       = acc_re[r];
        out[256 + (r + 8 * lh) * 16 + fl] = acc_im[r];
    }
}

// ---------------------------------------------------------------------------
// Kernel 2: reduce K-slices, apply a_weights (16x16) in f32, scale, pack
// complex64 output (b, f, m) -> interleaved re/im floats. Deterministic.
// ---------------------------------------------------------------------------
__global__ __launch_bounds__(256)
void spectral_reduce_project(const float* __restrict__ ws,
                             const float* __restrict__ aw,   // a_weights 16x16
                             float* __restrict__ out,
                             int S, float inv_sqrt_n)
{
    const int t = (int)(blockIdx.x * blockDim.x + threadIdx.x);
    if (t >= B_ * FDIM * COUT) return;

    const int m    = t & 15;
    const int f    = (t >> 4) & (FDIM - 1);
    const int b    = t >> 12;               // t / (256*16)
    const int tile = b * 16 + (f >> 4);
    const int frow = f & 15;

    float wcol[CIN];
    #pragma unroll
    for (int i = 0; i < CIN; ++i) wcol[i] = aw[i * COUT + m];

    float re = 0.0f, im = 0.0f;
    for (int s = 0; s < S; ++s) {
        const float* p = ws + (size_t)(tile * S + s) * 2 * 256 + frow * 16;
        #pragma unroll
        for (int i = 0; i < CIN; ++i) {
            re += p[i] * wcol[i];
            im += p[256 + i] * wcol[i];
        }
    }
    out[2 * t + 0] =  re * inv_sqrt_n;
    out[2 * t + 1] = -im * inv_sqrt_n;      // h_im carries the minus sign
}

// ---------------------------------------------------------------------------
extern "C" void kernel_launch(void* const* d_in, const int* in_sizes, int n_in,
                              void* d_out, int out_size, void* d_ws, size_t ws_size,
                              hipStream_t stream)
{
    (void)in_sizes; (void)n_in; (void)out_size;
    const float* a  = (const float*)d_in[0];   // (2,48,48,48,16)
    const float* x  = (const float*)d_in[1];   // (2,48,48,48,3)
    const float* aw = (const float*)d_in[2];   // (16,16)
    const float* xw = (const float*)d_in[3];   // (3,256)
    float* out = (float*)d_out;                // (2,256,16) complex64 re/im pairs

    // ws layout: [a_sw f16 swizzle (6.75 MB)] [split-K partials (S * 64 KB)]
    const size_t per_slice = (size_t)NTILES * 2 * 256 * sizeof(float); // 64 KB
    const int use_swz = (ws_size >= A_SW_BYTES + per_slice) ? 1 : 0;

    _Float16* a_sw    = (_Float16*)d_ws;
    float*    partial = use_swz ? (float*)((char*)d_ws + A_SW_BYTES)
                                : (float*)d_ws;
    const size_t avail = use_swz ? ws_size - A_SW_BYTES : ws_size;

    int S = (int)(avail / per_slice);
    if (S < 1)   S = 1;          // assume >= 64 KB of scratch
    if (S > 128) S = 128;        // 128 slices -> 4096 waves, 8 MB partials
    const int cps = (NCHUNKS + S - 1) / S;   // K-chunks per slice

    const int num_waves = NTILES * S;
    const int blocks1 = (num_waves * 32 + 255) / 256;

    if (use_swz) {
        const int n_oct = (int)(A_SW_ELEMS / 8);          // 442,368 octets
        swizzle_a_f16<<<(n_oct + 255) / 256, 256, 0, stream>>>(a, a_sw);
        spectral_wmma_partial<1><<<blocks1, 256, 0, stream>>>(
            a, a_sw, x, xw, partial, S, cps);
    } else {
        spectral_wmma_partial<0><<<blocks1, 256, 0, stream>>>(
            a, a_sw, x, xw, partial, S, cps);
    }

    const float inv_sqrt_n = 1.0f / sqrtf((float)NVOX);
    const int total2 = B_ * FDIM * COUT;     // 8192
    const int blocks2 = (total2 + 255) / 256;
    spectral_reduce_project<<<blocks2, 256, 0, stream>>>(
        partial, aw, out, S, inv_sqrt_n);
}
